// FP_Module_214748365417
// MI455X (gfx1250) — matrix operand order repaired and checked by reading, not compiled, not run on hardware
//
#include <hip/hip_runtime.h>

// Problem constants (match reference)
#define BB    8
#define NT    8192
#define NS    2048
#define C_TGT 128
#define C_SRC 256
#define C_IN  384
#define C_HID 256
#define C_OUT 128

typedef __attribute__((ext_vector_type(16))) __bf16 v16bf;
typedef __attribute__((ext_vector_type(8)))  float  v8f;

// K-offset inside a 16-bit 16x32 WMMA *A* fragment (CDNA5 ISA 7.12.2):
// lanes 0-15: K = {0..7} in V0..3 (2/VGPR), {16..23} in V4..7
// lanes 16-31: same +8.  e = element index in v16bf (v = e>>1, h = e&1).
__device__ __forceinline__ int klocalA(int e, int hi) {
    int v = e >> 1, h = e & 1;
    return ((v & 3) << 1) + h + ((v >> 2) << 4) + (hi << 3);
}

// ---------------------------------------------------------------------------
// Pack a row-major fp32 weight matrix W[K][N] into bf16 *B*-fragments.
// B-matrix layout (CDNA5 ISA 7.12.4/7.12.5 pattern, 32x16 16-bit):
//   lanes 0-15 hold K=0..7,16..23 (2 per VGPR), lanes 16-31 hold K=8..15,24..31,
//   N = lane & 15.  => per-lane element e maps to K = kbase + hi*8 + (e&7) + (e>>3)*16.
// Each lane's 16 elements are stored contiguously (32B -> 2x b128 loads).
// Fragment order: frag = kt * (N/16) + nt ; within fragment: lane-major.
// ---------------------------------------------------------------------------
__global__ void pack_w(const float* __restrict__ W, __bf16* __restrict__ out,
                       int K, int N) {
    int t = blockIdx.x * blockDim.x + threadIdx.x;
    int ntiles = N >> 4;
    int total = (K >> 5) * ntiles * 32;
    if (t >= total) return;
    int lane = t & 31;
    int frag = t >> 5;
    int kt = frag / ntiles, nt = frag % ntiles;
    int n  = (nt << 4) + (lane & 15);
    int hi = lane >> 4;
    __bf16* dst = out + ((size_t)frag * 32 + lane) * 16;
#pragma unroll
    for (int e = 0; e < 16; ++e) {
        int k = (kt << 5) + (hi << 3) + (e & 7) + ((e >> 3) << 4);   // table-order B layout (matches klocalA)
        dst[e] = (__bf16)W[(size_t)k * N + n];
    }
}

// ---------------------------------------------------------------------------
// KNN(K=3, inverse-squared-distance) + build bf16 combined = [x_tgt | interp].
// One block = 256 consecutive targets of one batch.
// LDS holds (x,y,z,|s|^2) per source point; inner loop evaluates the shifted
// distance d' = -2 t.s + |s|^2 (same ordering as true d^2) with 3 FMAs and a
// single broadcast ds_load_b128; |t|^2 is added to the 3 winners at the end.
// ---------------------------------------------------------------------------
__global__ __launch_bounds__(256) void knn_combined(
    const float* __restrict__ xt, const float* __restrict__ pt,
    const float* __restrict__ xs, const float* __restrict__ ps,
    __bf16* __restrict__ combined) {
    __shared__ float4 spk[NS];            // 32 KB: x, y, z, |s|^2
    __shared__ float  sw[256][3];
    __shared__ int    sidx[256][3];

    int tid   = threadIdx.x;
    int b     = blockIdx.x >> 5;          // 32 blocks of 256 targets per batch
    int tbase = blockIdx.x << 8;
    int t     = tbase + tid;

    const float* psrc = ps + (size_t)b * NS * 3;
    for (int i = tid; i < NS; i += 256) {
        float x = psrc[i * 3 + 0];
        float y = psrc[i * 3 + 1];
        float z = psrc[i * 3 + 2];
        spk[i] = make_float4(x, y, z, fmaf(x, x, fmaf(y, y, z * z)));
    }
    __syncthreads();

    float txv = pt[(size_t)t * 3 + 0];
    float tyv = pt[(size_t)t * 3 + 1];
    float tzv = pt[(size_t)t * 3 + 2];
    float tx2 = -2.f * txv, ty2 = -2.f * tyv, tz2 = -2.f * tzv;
    float tn  = fmaf(txv, txv, fmaf(tyv, tyv, tzv * tzv));

    float d0 = 3.4e38f, d1 = 3.4e38f, d2 = 3.4e38f;
    int   i0 = 0, i1 = 0, i2 = 0;
    for (int s = 0; s < NS; ++s) {
        float4 sp = spk[s];
        float d = fmaf(tx2, sp.x, fmaf(ty2, sp.y, fmaf(tz2, sp.z, sp.w)));
        if (d < d2) {
            if (d < d1) {
                if (d < d0) { d2 = d1; i2 = i1; d1 = d0; i1 = i0; d0 = d; i0 = s; }
                else        { d2 = d1; i2 = i1; d1 = d;  i1 = s; }
            } else          { d2 = d;  i2 = s; }
        }
    }
    // Recover true squared distances, inverse-distance weights (eps-clamped).
    float w0 = 1.f / fmaxf(d0 + tn, 1e-16f);
    float w1 = 1.f / fmaxf(d1 + tn, 1e-16f);
    float w2 = 1.f / fmaxf(d2 + tn, 1e-16f);
    float inv = 1.f / (w0 + w1 + w2);
    sw[tid][0] = w0 * inv; sw[tid][1] = w1 * inv; sw[tid][2] = w2 * inv;
    int sbase = b * NS;
    sidx[tid][0] = sbase + i0; sidx[tid][1] = sbase + i1; sidx[tid][2] = sbase + i2;
    __syncthreads();

    // Cooperative, lane-coalesced interpolation + write of combined rows.
    int wave = tid >> 5, lane = tid & 31;
    for (int j = wave; j < 256; j += 8) {
        int tr = tbase + j;
        __bf16* crow = combined + (size_t)tr * C_IN;
        const float* xtr = xt + (size_t)tr * C_TGT;
        for (int c = lane; c < C_TGT; c += 32) crow[c] = (__bf16)xtr[c];
        float a0 = sw[j][0], a1 = sw[j][1], a2 = sw[j][2];
        const float* s0 = xs + (size_t)sidx[j][0] * C_SRC;
        const float* s1 = xs + (size_t)sidx[j][1] * C_SRC;
        const float* s2 = xs + (size_t)sidx[j][2] * C_SRC;
        for (int c = lane; c < C_SRC; c += 32) {
            float v = a0 * s0[c] + a1 * s1[c] + a2 * s2[c];
            crow[C_TGT + c] = (__bf16)v;
        }
    }
}

// ---------------------------------------------------------------------------
// Fused MLP: out = relu( relu(A@W1+b1)@W2 + b2 + A@Ws + bs )
// 8 waves/block; each wave owns one 16-row M-tile (block covers 128 rows).
// A fragments loaded once, reused for GEMM1 and the shortcut GEMM.
// h tile bounced through LDS to convert C/D layout -> A layout (same wave).
// ---------------------------------------------------------------------------
__global__ __launch_bounds__(256) void fused_mlp(
    const __bf16* __restrict__ A,     // combined [65536 x 384] bf16
    const __bf16* __restrict__ W1p,   // packed fragments 12 kt x 16 nt
    const __bf16* __restrict__ W2p,   // packed fragments  8 kt x  8 nt
    const __bf16* __restrict__ Wsp,   // packed fragments 12 kt x  8 nt
    const float* __restrict__ b1, const float* __restrict__ b2,
    const float* __restrict__ bs, float* __restrict__ out) {
    __shared__ __bf16 hbuf[8][16][C_HID];   // 64 KB

    int tid  = threadIdx.x;
    int wave = tid >> 5, lane = tid & 31;
    int hi   = lane >> 4, lm = lane & 15;
    int m0   = blockIdx.x * 128 + wave * 16;

    // Load the wave's 12 A-fragments (K = 384) once.
    v16bf a[12];
#pragma unroll
    for (int kt = 0; kt < 12; ++kt) {
        const __bf16* arow = A + (size_t)(m0 + lm) * C_IN + kt * 32;
#pragma unroll
        for (int e = 0; e < 16; e += 2) {
            int k = klocalA(e, hi);
            a[kt][e]     = arow[k];
            a[kt][e + 1] = arow[k + 1];
        }
    }

    // GEMM1: h = relu(A @ W1 + b1) -> LDS (bf16, row-major per wave tile)
    for (int nt = 0; nt < 16; ++nt) {
        v8f acc = {};
#pragma unroll
        for (int kt = 0; kt < 12; ++kt) {
            v16bf bf = *(const v16bf*)(W1p + ((size_t)(kt * 16 + nt) * 32 + lane) * 16);
            acc = __builtin_amdgcn_wmma_f32_16x16x32_bf16(
                false, a[kt], false, bf, (short)0, acc, false, false);
        }
        float bias = b1[nt * 16 + lm];
#pragma unroll
        for (int r = 0; r < 8; ++r) {
            float v = acc[r] + bias;
            v = v > 0.f ? v : 0.f;
            // C/D layout: row = r + 8*hi, col = lm  (within this N-tile)
            hbuf[wave][r + 8 * hi][nt * 16 + lm] = (__bf16)v;
        }
    }

    // Re-load h as A-fragments (same-wave DS producer/consumer; in-order).
    v16bf ha[8];
#pragma unroll
    for (int kt = 0; kt < 8; ++kt) {
        const __bf16* hrow = &hbuf[wave][lm][kt * 32];
#pragma unroll
        for (int e = 0; e < 16; e += 2) {
            int k = klocalA(e, hi);
            ha[kt][e]     = hrow[k];
            ha[kt][e + 1] = hrow[k + 1];
        }
    }

    // GEMM2 + shortcut, fused accumulation; bias + ReLU; store.
    for (int nt = 0; nt < 8; ++nt) {
        v8f acc = {};
#pragma unroll
        for (int kt = 0; kt < 12; ++kt) {
            v16bf bf = *(const v16bf*)(Wsp + ((size_t)(kt * 8 + nt) * 32 + lane) * 16);
            acc = __builtin_amdgcn_wmma_f32_16x16x32_bf16(
                false, a[kt], false, bf, (short)0, acc, false, false);
        }
#pragma unroll
        for (int kt = 0; kt < 8; ++kt) {
            v16bf bf = *(const v16bf*)(W2p + ((size_t)(kt * 8 + nt) * 32 + lane) * 16);
            acc = __builtin_amdgcn_wmma_f32_16x16x32_bf16(
                false, ha[kt], false, bf, (short)0, acc, false, false);
        }
        float bias = b2[nt * 16 + lm] + bs[nt * 16 + lm];
#pragma unroll
        for (int r = 0; r < 8; ++r) {
            float v = acc[r] + bias;
            out[(size_t)(m0 + r + 8 * hi) * C_OUT + nt * 16 + lm] = v > 0.f ? v : 0.f;
        }
    }
}

// ---------------------------------------------------------------------------
extern "C" void kernel_launch(void* const* d_in, const int* in_sizes, int n_in,
                              void* d_out, int out_size, void* d_ws, size_t ws_size,
                              hipStream_t stream) {
    const float* xt = (const float*)d_in[0];
    const float* pt = (const float*)d_in[1];
    const float* xs = (const float*)d_in[3];
    const float* ps = (const float*)d_in[4];
    const float* W1 = (const float*)d_in[6];
    const float* b1 = (const float*)d_in[7];
    const float* W2 = (const float*)d_in[8];
    const float* b2 = (const float*)d_in[9];
    const float* Ws = (const float*)d_in[10];
    const float* bs = (const float*)d_in[11];

    char* ws = (char*)d_ws;
    __bf16* combined = (__bf16*)ws;                         // 65536*384*2 = 50,331,648 B
    size_t off = (size_t)65536 * C_IN * 2;
    __bf16* W1p = (__bf16*)(ws + off); off += (size_t)C_IN * C_HID * 2;   // 196,608 B
    __bf16* W2p = (__bf16*)(ws + off); off += (size_t)C_HID * C_OUT * 2;  //  65,536 B
    __bf16* Wsp = (__bf16*)(ws + off);                                     //  98,304 B

    // Pack weights into WMMA B-fragment layout (bf16).
    pack_w<<<((C_IN  / 32) * (C_HID / 16) * 32 + 255) / 256, 256, 0, stream>>>(W1, W1p, C_IN,  C_HID);
    pack_w<<<((C_HID / 32) * (C_OUT / 16) * 32 + 255) / 256, 256, 0, stream>>>(W2, W2p, C_HID, C_OUT);
    pack_w<<<((C_IN  / 32) * (C_OUT / 16) * 32 + 255) / 256, 256, 0, stream>>>(Ws, Wsp, C_IN,  C_OUT);

    // KNN + combined-feature build (bf16).
    knn_combined<<<(BB * NT) / 256, 256, 0, stream>>>(xt, pt, xs, ps, combined);

    // Fused two-layer MLP + shortcut on WMMA.
    fused_mlp<<<(BB * NT) / 128, 256, 0, stream>>>(combined, W1p, W2p, Wsp,
                                                   b1, b2, bs, (float*)d_out);
}
